// DecoderRNNsearch_68685116997921
// MI455X (gfx1250) — matrix operand order repaired and checked by reading, compiled, not run on hardware
//
#include <hip/hip_runtime.h>
#include <hip/hip_bf16.h>

// ---------------------------------------------------------------------------
// DecoderRNNsearch for MI455X (gfx1250, wave32, WMMA + TDM).
//   B=64, T=64, L=64, EMB=512, HID=1024, CTXD=2048, 3H=3072, VOCAB=32000
// bf16 WMMA GEMMs with fp32 accumulate; A/B K-tiles staged in LDS by the
// Tensor Data Mover (tensor_load_to_lds, TENSORcnt), double buffered so the
// DMA of stage k+1 overlaps the WMMAs of stage k. TDM LDS padding gives a
// 144B row pitch -> conflict-free ds_load_b128 fragment reads.
// ---------------------------------------------------------------------------

typedef __attribute__((ext_vector_type(16))) __bf16 v16bf;
typedef __attribute__((ext_vector_type(8)))  float  v8f;
typedef __attribute__((ext_vector_type(4)))  unsigned int u32x4;
typedef __attribute__((ext_vector_type(8)))  int i32x8;
typedef __attribute__((ext_vector_type(4)))  int i32x4;

#define KB   64      // batch
#define KT   64      // target steps
#define KL   64      // context length
#define KH   1024    // hidden
#define KC   2048    // ctx dim
#define KE   512     // embedding dim
#define KG   3072    // 3*H

__device__ __forceinline__ __bf16 f2bf(float f) {
  unsigned u = __builtin_bit_cast(unsigned, f);
  unsigned r = (u + 0x7FFFu + ((u >> 16) & 1u)) >> 16;   // RNE
  unsigned short s = (unsigned short)r;
  return __builtin_bit_cast(__bf16, s);
}

// -------------------- elementwise converts / gathers -----------------------

__global__ void convert_f32_bf16(const float* __restrict__ src, long ld_src,
                                 long col0, __bf16* __restrict__ dst,
                                 long ld_dst, int rows, int cols) {
  long i = (long)blockIdx.x * blockDim.x + threadIdx.x;
  long total = (long)rows * cols;
  if (i >= total) return;
  long r = i / cols;
  long c = i - r * cols;
  dst[r * ld_dst + c] = f2bf(src[r * ld_src + col0 + c]);
}

__global__ void emb_gather_bf16(const int* __restrict__ tgt,
                                const float* __restrict__ table,
                                __bf16* __restrict__ dst) {
  long i = (long)blockIdx.x * blockDim.x + threadIdx.x;   // KB*KT*KE total
  int rt = (int)(i >> 9);          // b*T + t
  int c  = (int)(i & 511);
  dst[i] = f2bf(table[(long)tgt[rt] * KE + c]);
}

__global__ void h_to_bf16(const float* __restrict__ hprev, long stride,
                          __bf16* __restrict__ dst) {
  int i = blockIdx.x * blockDim.x + threadIdx.x;          // KB*KH total
  int b = i >> 10, j = i & 1023;
  dst[i] = f2bf(hprev[(long)b * stride + j]);
}

// -------------------- TDM tile load (2D, bf16 rows) ------------------------
// Loads `rows` rows of 64 bf16 (32 dwords) from global (row pitch ld_elems
// bf16) into LDS at lds_byte, with 4 dwords of LDS padding after every row
// (pad_interval = 32 dwords) -> LDS row pitch = 72 bf16 = 144 bytes.
// Descriptor per CDNA5 ISA 8.3/8.4; groups 2/3/4 zero (2D tile).
// NOTE: this toolchain declares the 6-arg builtin:
//   (u32x4 g0, i32x8 g1, i32x4, i32x4, i32x8, i32 cpol)

__device__ __forceinline__ void tdm_load_tile64(unsigned lds_byte,
                                                const __bf16* gsrc,
                                                int rows, int ld_elems) {
  unsigned long long ga = (unsigned long long)(const void*)gsrc;
  const int tile_w_dw  = 32;                 // 64 bf16 per row
  const int stride_dw  = ld_elems >> 1;      // row pitch in dwords

  u32x4 g0;
  g0[0] = 1u;                                          // count=1, user mode
  g0[1] = lds_byte;                                    // lds_addr
  g0[2] = (unsigned)(ga & 0xFFFFFFFFull);              // global_addr[31:0]
  g0[3] = (unsigned)((ga >> 32) & 0x01FFFFFFull)       // global_addr[56:32]
        | (2u << 30);                                  // type=2 (image)

  i32x8 g1;
  g1[0] = (2 << 16)                                    // data_size = 4B
        | (1 << 20)                                    // pad_enable
        | (4 << 22)                                    // pad_interval: 32 dw
        | (3 << 25);                                   // pad_amount: 4 dw
  g1[1] = (tile_w_dw & 0xFFFF) << 16;                  // tensor_dim0[15:0]
  g1[2] = ((tile_w_dw >> 16) & 0xFFFF)                 // tensor_dim0[31:16]
        | ((rows & 0xFFFF) << 16);                     // tensor_dim1[15:0]
  g1[3] = ((rows >> 16) & 0xFFFF)                      // tensor_dim1[31:16]
        | ((tile_w_dw & 0xFFFF) << 16);                // tile_dim0
  g1[4] = (rows & 0xFFFF);                             // tile_dim1, dim2=0
  g1[5] = stride_dw;                                   // dim0_stride[31:0]
  g1[6] = 0;                                           // dim0_stride[47:32]
  g1[7] = 0;

  i32x4 z4 = {0, 0, 0, 0};
  i32x8 z8 = {0, 0, 0, 0, 0, 0, 0, 0};
  __builtin_amdgcn_tensor_load_to_lds(g0, g1, z4, z4, z8, 0);
}

// -------------------- WMMA bf16 NT GEMM (TDM + LDS double buffer) ----------
// C[M,N] = A[M,K](bf16,rm) * B[N,K](bf16,rm)^T  (+ bias + Cin)
// Block: 256 threads = 8 waves (4 M-subtiles x 2 N-halves), tile 64 x 128.
// K stage = 64 (2 WMMA k-chunks). M%64==0, N%128==0, K%64==0 at all call
// sites -> no wave ever diverges around barriers; EXEC all-ones for WMMA.
// A/B fragments: contiguous k = 16*half + s (same in A and B); C layout per ISA 7.12.2.

#define LDSP 72      // padded LDS row pitch in bf16 (set by TDM padding)

union FragAB { v16bf bf; uint4 q[2]; };

__global__ __launch_bounds__(256)
void wmma_gemm_nt(const __bf16* __restrict__ A, const __bf16* __restrict__ Bm,
                  const float* __restrict__ Cin, long ldc_in,
                  const float* __restrict__ bias,
                  float* __restrict__ Cout, long ldc_out,
                  int M, int N, int K, int lda, int ldb) {
  __shared__ __bf16 sA[2][64][LDSP];    // 18 KB
  __shared__ __bf16 sB[2][128][LDSP];   // 36 KB

  const int lane = threadIdx.x & 31;
  const int wave = threadIdx.x >> 5;
  const int wm   = wave & 3;                 // M sub-tile within block
  const int wn   = wave >> 2;                // N half within block
  const int m0   = blockIdx.y * 64;
  const int n0   = blockIdx.x * 128;
  const int half = lane >> 4;
  const int sub  = lane & 15;

  v8f acc[4] = {};

  if (wave == 0) {
    tdm_load_tile64((unsigned)(uintptr_t)&sA[0][0][0],
                    A + (long)m0 * lda, 64, lda);
    tdm_load_tile64((unsigned)(uintptr_t)&sB[0][0][0],
                    Bm + (long)n0 * ldb, 128, ldb);
  }

  int cur = 0;
  for (int k0 = 0; k0 < K; k0 += 64) {
    const bool has_next = (k0 + 64 < K);
    if (wave == 0) {
      if (has_next) {
        tdm_load_tile64((unsigned)(uintptr_t)&sA[cur ^ 1][0][0],
                        A + (long)m0 * lda + k0 + 64, 64, lda);
        tdm_load_tile64((unsigned)(uintptr_t)&sB[cur ^ 1][0][0],
                        Bm + (long)n0 * ldb + k0 + 64, 128, ldb);
        __builtin_amdgcn_s_wait_tensorcnt(2);   // current stage complete
      } else {
        __builtin_amdgcn_s_wait_tensorcnt(0);
      }
    }
    __syncthreads();                            // publish LDS stage

#pragma unroll
    for (int kk = 0; kk < 64; kk += 32) {
      FragAB a, b0, b1, b2, b3;
      a.q[0]  = *(const uint4*)&sA[cur][wm * 16 + sub][kk + 16 * half];
      a.q[1]  = *(const uint4*)&sA[cur][wm * 16 + sub][kk + 16 * half + 8];
      b0.q[0] = *(const uint4*)&sB[cur][wn * 64 +  0 + sub][kk + 16 * half];
      b0.q[1] = *(const uint4*)&sB[cur][wn * 64 +  0 + sub][kk + 16 * half + 8];
      b1.q[0] = *(const uint4*)&sB[cur][wn * 64 + 16 + sub][kk + 16 * half];
      b1.q[1] = *(const uint4*)&sB[cur][wn * 64 + 16 + sub][kk + 16 * half + 8];
      b2.q[0] = *(const uint4*)&sB[cur][wn * 64 + 32 + sub][kk + 16 * half];
      b2.q[1] = *(const uint4*)&sB[cur][wn * 64 + 32 + sub][kk + 16 * half + 8];
      b3.q[0] = *(const uint4*)&sB[cur][wn * 64 + 48 + sub][kk + 16 * half];
      b3.q[1] = *(const uint4*)&sB[cur][wn * 64 + 48 + sub][kk + 16 * half + 8];
      acc[0] = __builtin_amdgcn_wmma_f32_16x16x32_bf16(
                   false, a.bf, false, b0.bf, (short)0, acc[0], false, false);
      acc[1] = __builtin_amdgcn_wmma_f32_16x16x32_bf16(
                   false, a.bf, false, b1.bf, (short)0, acc[1], false, false);
      acc[2] = __builtin_amdgcn_wmma_f32_16x16x32_bf16(
                   false, a.bf, false, b2.bf, (short)0, acc[2], false, false);
      acc[3] = __builtin_amdgcn_wmma_f32_16x16x32_bf16(
                   false, a.bf, false, b3.bf, (short)0, acc[3], false, false);
    }
    __syncthreads();                            // done reading before reuse
    cur ^= 1;
  }

#pragma unroll
  for (int j = 0; j < 4; ++j) {
    int col = n0 + wn * 64 + j * 16 + sub;
    float bv = bias ? bias[col] : 0.0f;
#pragma unroll
    for (int v = 0; v < 8; ++v) {
      int row = m0 + wm * 16 + v + 8 * half;
      float r = acc[j][v] + bv;
      if (Cin) r += Cin[(long)row * ldc_in + col];
      Cout[(long)row * ldc_out + col] = r;
    }
  }
}

// -------------------- attention (scores, softmax, context) -----------------
// One block per batch element b. 256 threads.

__global__ __launch_bounds__(256)
void attention_kernel(const float* __restrict__ hprev, long h_stride,
                      const float* __restrict__ ctxW,   // [B*L, H]
                      const float* __restrict__ ctx,    // [B*L, C]
                      const int* __restrict__ lens,
                      __bf16* __restrict__ ct_bf) {     // [B, C]
  __shared__ float s_part[256];
  __shared__ float s_attn[KL];
  __shared__ float s_red[KL];

  const int b   = blockIdx.x;
  const int tid = threadIdx.x;
  const int l   = tid & 63;
  const int part = tid >> 6;                 // 4 partials of 256 h each

  const float* hb = hprev + (long)b * h_stride;
  const float* cw = ctxW + ((long)b * KL + l) * KH + part * 256;
  float acc = 0.0f;
#pragma unroll 8
  for (int h = 0; h < 256; ++h) acc += hb[part * 256 + h] * cw[h];
  s_part[tid] = acc;
  __syncthreads();

  if (tid < KL) {
    float sc = s_part[tid] + s_part[64 + tid] + s_part[128 + tid]
             + s_part[192 + tid];
    sc = (tid < lens[b]) ? sc : -1.0e9f;
    s_attn[tid] = sc;
    s_red[tid]  = sc;
  }
  __syncthreads();

  for (int s = 32; s >= 1; s >>= 1) {
    if (tid < s) s_red[tid] = fmaxf(s_red[tid], s_red[tid + s]);
    __syncthreads();
  }
  float mx = s_red[0];
  __syncthreads();

  if (tid < KL) {
    float e = __expf(s_attn[tid] - mx);
    s_attn[tid] = e;
    s_red[tid]  = e;
  }
  __syncthreads();

  for (int s = 32; s >= 1; s >>= 1) {
    if (tid < s) s_red[tid] += s_red[tid + s];
    __syncthreads();
  }
  float inv = 1.0f / s_red[0];

  const float* cb = ctx + (long)b * KL * KC;
  for (int c = tid; c < KC; c += 256) {
    float a2 = 0.0f;
#pragma unroll 8
    for (int l2 = 0; l2 < KL; ++l2) a2 += s_attn[l2] * cb[(long)l2 * KC + c];
    ct_bf[(long)b * KC + c] = f2bf(a2 * inv);
  }
}

// -------------------- fused GRU gate update --------------------------------

__global__ __launch_bounds__(256)
void gru_gate_kernel(const float* __restrict__ gx,   // [B, 3H]
                     const float* __restrict__ gh,   // [B, 3H]
                     const float* __restrict__ hprev, long h_stride,
                     float* __restrict__ out_t,      // outs + t*H
                     float* __restrict__ h_last, int is_last) {
  int i = blockIdx.x * blockDim.x + threadIdx.x;     // KB*KH total
  int b = i >> 10, j = i & 1023;
  long g = (long)b * KG;
  float r = 1.0f / (1.0f + __expf(-(gx[g + j]           + gh[g + j])));
  float z = 1.0f / (1.0f + __expf(-(gx[g + KH + j]      + gh[g + KH + j])));
  float n = tanhf(gx[g + 2 * KH + j] + r * gh[g + 2 * KH + j]);
  float hp = hprev[(long)b * h_stride + j];
  float hn = (1.0f - z) * n + z * hp;
  out_t[(long)b * (KT * KH) + j] = hn;
  if (is_last) h_last[(long)b * KH + j] = hn;
}

// -------------------- host orchestration -----------------------------------

static inline unsigned cdiv(long a, long b) { return (unsigned)((a + b - 1) / b); }

extern "C" void kernel_launch(void* const* d_in, const int* in_sizes, int n_in,
                              void* d_out, int out_size, void* d_ws, size_t ws_size,
                              hipStream_t stream) {
  const int*   tgt   = (const int*)d_in[0];
  const float* ctx   = (const float*)d_in[1];
  const float* h0    = (const float*)d_in[2];
  const int*   lens  = (const int*)d_in[3];
  const float* table = (const float*)d_in[4];
  const float* W_a   = (const float*)d_in[5];
  const float* W_ih  = (const float*)d_in[6];
  const float* W_hh  = (const float*)d_in[7];
  const float* b_ih  = (const float*)d_in[8];
  const float* b_hh  = (const float*)d_in[9];

  float* outs   = (float*)d_out;                       // [B, T, H]
  float* h_last = outs + (size_t)KB * KT * KH;         // [B, H]

  char* w = (char*)d_ws;
  auto carve = [&](size_t bytes) -> void* {
    void* p = (void*)w;
    w += (bytes + 255) & ~(size_t)255;
    return p;
  };
  __bf16* ctx_bf   = (__bf16*)carve((size_t)KB * KL * KC * 2);
  __bf16* Wa_bf    = (__bf16*)carve((size_t)KH * KC * 2);
  __bf16* Wie_bf   = (__bf16*)carve((size_t)KG * KE * 2);   // W_ih[:, :E]
  __bf16* Wic_bf   = (__bf16*)carve((size_t)KG * KC * 2);   // W_ih[:, E:]
  __bf16* Whh_bf   = (__bf16*)carve((size_t)KG * KH * 2);
  __bf16* emb_bf   = (__bf16*)carve((size_t)KB * KT * KE * 2);
  float*  ctxW     = (float*)carve((size_t)KB * KL * KH * 4);
  float*  gx_pre   = (float*)carve((size_t)KB * KT * KG * 4);
  __bf16* ct_bf    = (__bf16*)carve((size_t)KB * KC * 2);
  __bf16* h_bf     = (__bf16*)carve((size_t)KB * KH * 2);
  float*  gx_step  = (float*)carve((size_t)KB * KG * 4);
  float*  gh_step  = (float*)carve((size_t)KB * KG * 4);

  // one-time bf16 conversions
  convert_f32_bf16<<<cdiv((long)KB * KL * KC, 256), 256, 0, stream>>>(
      ctx, KC, 0, ctx_bf, KC, KB * KL, KC);
  convert_f32_bf16<<<cdiv((long)KH * KC, 256), 256, 0, stream>>>(
      W_a, KC, 0, Wa_bf, KC, KH, KC);
  convert_f32_bf16<<<cdiv((long)KG * KE, 256), 256, 0, stream>>>(
      W_ih, KE + KC, 0, Wie_bf, KE, KG, KE);
  convert_f32_bf16<<<cdiv((long)KG * KC, 256), 256, 0, stream>>>(
      W_ih, KE + KC, KE, Wic_bf, KC, KG, KC);
  convert_f32_bf16<<<cdiv((long)KG * KH, 256), 256, 0, stream>>>(
      W_hh, KH, 0, Whh_bf, KH, KG, KH);
  emb_gather_bf16<<<cdiv((long)KB * KT * KE, 256), 256, 0, stream>>>(
      tgt, table, emb_bf);

  // hoisted GEMMs (big M = 4096)
  wmma_gemm_nt<<<dim3(KH / 128, (KB * KL) / 64), 256, 0, stream>>>(
      ctx_bf, Wa_bf, nullptr, 0, nullptr, ctxW, KH,
      KB * KL, KH, KC, KC, KC);
  wmma_gemm_nt<<<dim3(KG / 128, (KB * KT) / 64), 256, 0, stream>>>(
      emb_bf, Wie_bf, nullptr, 0, b_ih, gx_pre, KG,
      KB * KT, KG, KE, KE, KE);

  // sequential scan over T
  for (int t = 0; t < KT; ++t) {
    const float* hprev;
    long hstride;
    if (t == 0) { hprev = h0;                          hstride = KH; }
    else        { hprev = outs + (size_t)(t - 1) * KH; hstride = (long)KT * KH; }

    attention_kernel<<<KB, 256, 0, stream>>>(hprev, hstride, ctxW, ctx, lens,
                                             ct_bf);
    h_to_bf16<<<cdiv((long)KB * KH, 256), 256, 0, stream>>>(hprev, hstride,
                                                            h_bf);
    // gx = gx_pre[:, t, :] + c_t . W_ih[:, E:]^T     M=64 N=3072 K=2048
    wmma_gemm_nt<<<dim3(KG / 128, 1), 256, 0, stream>>>(
        ct_bf, Wic_bf, gx_pre + (size_t)t * KG, (long)KT * KG, nullptr,
        gx_step, KG, KB, KG, KC, KC, KC);
    // gh = h . W_hh^T + b_hh                         M=64 N=3072 K=1024
    wmma_gemm_nt<<<dim3(KG / 128, 1), 256, 0, stream>>>(
        h_bf, Whh_bf, nullptr, 0, b_hh, gh_step, KG,
        KB, KG, KH, KH, KH);

    gru_gate_kernel<<<cdiv((long)KB * KH, 256), 256, 0, stream>>>(
        gx_step, gh_step, hprev, hstride, outs + (size_t)t * KH, h_last,
        (t == KT - 1) ? 1 : 0);
  }
}
